// CustomAutoencoder_32736240730469
// MI455X (gfx1250) — matrix operand order repaired and checked by reading, not compiled, not run on hardware
//
#include <hip/hip_runtime.h>
#include <math.h>

typedef __attribute__((ext_vector_type(16))) __bf16 v16bf;
typedef __attribute__((ext_vector_type(8)))  __bf16 v8bf;
typedef __attribute__((ext_vector_type(8)))  float  v8f;
typedef __attribute__((ext_vector_type(4)))  unsigned int v4u;
typedef __attribute__((ext_vector_type(8)))  int v8i;
typedef __attribute__((ext_vector_type(4)))  int v4i;

#define ACT_RELU 0
#define ACT_SIG  1

#define BM 128
#define BN 128
#define BK 32
// LDS: double-buffered A (128x32 bf16 = 8KB each) then B (128x32 bf16 = 8KB each)
#define LDS_A0 0u
#define LDS_A1 8192u
#define LDS_B0 16384u
#define LDS_B1 24576u
#define LDS_TOTAL 32768u

union BF16x16 { v16bf v; v8bf h[2]; };

// ---------------------------------------------------------------------------
// Prep kernels (tiny, once per launch): bf16-convert / transpose / pad / scatter
// ---------------------------------------------------------------------------

// Generic: out[Npad][Kpad] = transpose of in[Krows][Ncols], zero-padded, bf16.
__global__ __launch_bounds__(256) void prep_transpose(const float* __restrict__ in,
                                                      __bf16* __restrict__ out,
                                                      int Krows, int Ncols,
                                                      int Kpad, int Npad) {
  int i = blockIdx.x * blockDim.x + threadIdx.x;
  if (i >= Kpad * Npad) return;
  int n = i / Kpad, k = i % Kpad;
  float v = (k < Krows && n < Ncols) ? in[(size_t)k * Ncols + n] : 0.0f;
  out[i] = (__bf16)v;
}

// Zero Wgt (bf16 [512 g][512 k], already N-major) and pad gb -> gbp[512] fp32.
__global__ __launch_bounds__(256) void prep_wg_zero(__bf16* __restrict__ Wgt,
                                                    float* __restrict__ gbp,
                                                    const float* __restrict__ gb, int G) {
  int i = blockIdx.x * blockDim.x + threadIdx.x;
  if (i < 512 * 512) Wgt[i] = (__bf16)0.0f;
  if (i < 512) gbp[i] = (i < G) ? gb[i] : 0.0f;
}

// Scatter gw*mask: Wgt[g][idx[g,k]] = gw[g,k]*mask[g,k]. idx unique per g.
__global__ __launch_bounds__(256) void prep_wg_scatter(__bf16* __restrict__ Wgt,
                                                       const float* __restrict__ gw,
                                                       const float* __restrict__ mask,
                                                       const int* __restrict__ idx,
                                                       int G, int K) {
  int i = blockIdx.x * blockDim.x + threadIdx.x;
  if (i >= G * K) return;
  float m = mask[i];
  if (m == 0.0f) return;
  int g = i / K;
  Wgt[(size_t)g * 512 + idx[i]] = (__bf16)(gw[i] * m);
}

// Vectorized fp32 -> bf16 for x.
__global__ __launch_bounds__(256) void prep_cvt_bf16(const float* __restrict__ in,
                                                     __bf16* __restrict__ out, int n4) {
  int i = blockIdx.x * blockDim.x + threadIdx.x;
  if (i >= n4) return;
  float4 v = ((const float4*)in)[i];
  out[4 * i + 0] = (__bf16)v.x;
  out[4 * i + 1] = (__bf16)v.y;
  out[4 * i + 2] = (__bf16)v.z;
  out[4 * i + 3] = (__bf16)v.w;
}

// ---------------------------------------------------------------------------
// TDM: DMA a 2D tile (rows x 32 bf16, row stride = stride_elts bf16) from
// global memory into LDS at byte offset lds_off. D# per CDNA5 ISA:
//   g0: count=1 | lds_addr | global_addr[56:0] | type=2
//   g1: data_size=1(2B), tensor_dim0=32, tensor_dim1=rows, tile_dim0=32,
//       tile_dim1=rows, tensor_dim0_stride=stride_elts
// 6-arg toolchain form: (v4u g0, v8i g1, v4i g2, v4i g3, v8i g4, i32 cpol).
// ---------------------------------------------------------------------------
__device__ __forceinline__ void tdm_load_tile(unsigned lds_off,
                                              const __bf16* gsrc,
                                              unsigned rows, unsigned stride_elts) {
  unsigned long long ga = (unsigned long long)(uintptr_t)gsrc;
  v4u g0;
  g0[0] = 1u;                                               // count=1 (valid D#)
  g0[1] = lds_off;                                          // lds_addr (bytes)
  g0[2] = (unsigned)(ga & 0xFFFFFFFFu);                     // global_addr[31:0]
  g0[3] = (unsigned)((ga >> 32) & 0x01FFFFFFu) | (2u << 30);// addr[56:32] | type=2
  v8i g1;
  g1[0] = (int)(1u << 16);                                  // data_size=1 (2 bytes)
  g1[1] = (int)((32u & 0xFFFFu) << 16);                     // tensor_dim0[15:0]=32
  g1[2] = (int)((rows & 0xFFFFu) << 16);                    // dim0[31:16]=0 | tensor_dim1[15:0]
  g1[3] = (int)((rows >> 16) | (32u << 16));                // tensor_dim1[31:16] | tile_dim0=32
  g1[4] = (int)(rows & 0xFFFFu);                            // tile_dim1=rows, tile_dim2=0
  g1[5] = (int)stride_elts;                                 // tensor_dim0_stride[31:0]
  g1[6] = 0;                                                // stride[47:32]=0 | dim1_stride=0
  g1[7] = 0;
  v4i z4; z4[0] = 0; z4[1] = 0; z4[2] = 0; z4[3] = 0;
  v8i z8; z8[0] = 0; z8[1] = 0; z8[2] = 0; z8[3] = 0;
          z8[4] = 0; z8[5] = 0; z8[6] = 0; z8[7] = 0;
  __builtin_amdgcn_tensor_load_to_lds(g0, g1, z4, z4, z8, 0);
}

// ---------------------------------------------------------------------------
// GEMM + bias + activation.  C[M x Ntot] = act(A[M x K] * Bt^T + bias)
// A: bf16 [M][K] row-major.  Bt: bf16 [Npad][K] (pre-transposed, zero-padded).
// 256 threads = 8 waves; block tile 128x128; each wave owns a 32x64 C-tile
// -> 8 WMMAs per K-step from 2 A-frags + 4 B-frags (1.5 ds_load_b128/WMMA).
// Tiles staged by TDM, double-buffered, issued by wave 0 only.
// ---------------------------------------------------------------------------
template <typename TOUT, int ACT>
__global__ __launch_bounds__(256) void gemm_tdm(const __bf16* __restrict__ A,
                                                const __bf16* __restrict__ Bt,
                                                const float* __restrict__ bias,
                                                TOUT* __restrict__ C,
                                                int Ktot, int Ntot) {
  extern __shared__ char smem[];
  const unsigned offA[2] = {LDS_A0, LDS_A1};
  const unsigned offB[2] = {LDS_B0, LDS_B1};

  const int tid  = threadIdx.x;
  const int lane = tid & 31;
  const int half = lane >> 4;   // lanes 0-15 vs 16-31
  const int mrow = lane & 15;
  const int wave = tid >> 5;
  const int wrow = (wave & 3) * 32;   // wave's 32-row strip within tile
  const int wcol = (wave >> 2) * 64;  // wave's 64-col strip within tile

  const int row0 = blockIdx.x * BM;
  const int col0 = blockIdx.y * BN;

  v8f acc[2][4] = {};

  // Prologue: DMA tile 0 into buffer 0.
  if (tid < 32) {
    tdm_load_tile(offA[0], A + (size_t)row0 * Ktot, BM, (unsigned)Ktot);
    tdm_load_tile(offB[0], Bt + (size_t)col0 * Ktot, BN, (unsigned)Ktot);
  }

  int buf = 0;
  for (int k0 = 0; k0 < Ktot; k0 += BK, buf ^= 1) {
    const bool more = (k0 + BK) < Ktot;
    if (tid < 32) {
      if (more) {
        // Issue tile i+1 into the other buffer, then wait for tile i (in-order).
        tdm_load_tile(offA[buf ^ 1], A + (size_t)row0 * Ktot + (k0 + BK), BM, (unsigned)Ktot);
        tdm_load_tile(offB[buf ^ 1], Bt + (size_t)col0 * Ktot + (k0 + BK), BN, (unsigned)Ktot);
        __builtin_amdgcn_s_wait_tensorcnt(2);
      } else {
        __builtin_amdgcn_s_wait_tensorcnt(0);
      }
    }
    __syncthreads();

    const char* pa = smem + offA[buf];
    const char* pb = smem + offB[buf];

    // CDNA5 16-bit WMMA fragment layouts (rows are 32 bf16 = 64 bytes in LDS):
    //  A: lanes 0-15 hold K {0..7,16..23}; lanes 16-31 hold K {8..15,24..31}
    //  B: lanes 0-15 hold K {0..7,16..23}; lanes 16-31 hold K {8..15,24..31}
    BF16x16 a[2], b[4];
#pragma unroll
    for (int m = 0; m < 2; ++m) {
      const int ra = wrow + m * 16 + mrow;
      a[m].h[0] = *(const v8bf*)(pa + ra * 64 + 16 * half);
      a[m].h[1] = *(const v8bf*)(pa + ra * 64 + 32 + 16 * half);
    }
#pragma unroll
    for (int s = 0; s < 4; ++s) {
      const int rb = wcol + s * 16 + mrow;
      b[s].h[0] = *(const v8bf*)(pb + rb * 64 + 16 * half);
      b[s].h[1] = *(const v8bf*)(pb + rb * 64 + 32 + 16 * half);
    }
#pragma unroll
    for (int m = 0; m < 2; ++m)
#pragma unroll
      for (int s = 0; s < 4; ++s)
        acc[m][s] = __builtin_amdgcn_wmma_f32_16x16x32_bf16(
            false, a[m].v, false, b[s].v, (short)0, acc[m][s], false, false);
    __syncthreads();
  }

  // Epilogue. C/D layout: VGPR r holds M = r + 8*half, N = lane%16.
#pragma unroll
  for (int s = 0; s < 4; ++s) {
    const int col = col0 + wcol + s * 16 + mrow;
    if (col >= Ntot) continue;
    const float bcol = bias[col];
#pragma unroll
    for (int m = 0; m < 2; ++m) {
#pragma unroll
      for (int r = 0; r < 8; ++r) {
        const int row = row0 + wrow + m * 16 + 8 * half + r;
        float v = acc[m][s][r] + bcol;
        v = (ACT == ACT_RELU) ? fmaxf(v, 0.0f) : 1.0f / (1.0f + expf(-v));
        C[(size_t)row * Ntot + col] = (TOUT)v;
      }
    }
  }
}

// ---------------------------------------------------------------------------
extern "C" void kernel_launch(void* const* d_in, const int* in_sizes, int n_in,
                              void* d_out, int out_size, void* d_ws, size_t ws_size,
                              hipStream_t stream) {
  (void)n_in; (void)out_size; (void)ws_size;

  const float* x    = (const float*)d_in[0];
  const float* gw   = (const float*)d_in[1];
  const float* gb   = (const float*)d_in[2];
  const float* w1   = (const float*)d_in[3];
  const float* b1   = (const float*)d_in[4];
  const float* w2   = (const float*)d_in[5];
  const float* b2   = (const float*)d_in[6];
  const float* dw1  = (const float*)d_in[7];
  const float* db1  = (const float*)d_in[8];
  const float* dw2  = (const float*)d_in[9];
  const float* db2  = (const float*)d_in[10];
  const float* dw3  = (const float*)d_in[11];
  const float* db3  = (const float*)d_in[12];
  const int*   idx  = (const int*)d_in[13];
  const float* mask = (const float*)d_in[14];

  const int SAMPLE = 512;
  const int G      = in_sizes[2];           // 510
  const int Kmax   = in_sizes[13] / G;      // neighbor table width
  const int BATCH  = in_sizes[0] / SAMPLE;  // 8192
  const int MT     = BATCH / BM;            // 64 row tiles

  // Workspace carve-up (bf16 transposed weights, bf16 activations).
  char* ws = (char*)d_ws;
  size_t off = 0;
  __bf16* Wgt  = (__bf16*)(ws + off); off += (size_t)512 * 512 * 2;  // [512 g][512 k]
  __bf16* w1t  = (__bf16*)(ws + off); off += (size_t)128 * 512 * 2;  // [128][512]
  __bf16* w2t  = (__bf16*)(ws + off); off += (size_t)128 * 128 * 2;  // [128][128] (N padded)
  __bf16* dw1t = (__bf16*)(ws + off); off += (size_t)128 * 32  * 2;  // [128][32]
  __bf16* dw2t = (__bf16*)(ws + off); off += (size_t)256 * 128 * 2;  // [256][128]
  __bf16* dw3t = (__bf16*)(ws + off); off += (size_t)512 * 256 * 2;  // [512][256]
  float*  gbp  = (float*) (ws + off); off += (size_t)512 * 4;
  off = (off + 255) & ~(size_t)255;
  __bf16* xb = (__bf16*)(ws + off); off += (size_t)BATCH * 512 * 2;
  __bf16* h1 = (__bf16*)(ws + off); off += (size_t)BATCH * 512 * 2;
  __bf16* h2 = (__bf16*)(ws + off); off += (size_t)BATCH * 128 * 2;
  __bf16* h3 = (__bf16*)(ws + off); off += (size_t)BATCH * 32  * 2;
  __bf16* h4 = (__bf16*)(ws + off); off += (size_t)BATCH * 128 * 2;
  __bf16* h5 = (__bf16*)(ws + off); off += (size_t)BATCH * 256 * 2;

  // --- Prep: pack weights (bf16, [N][K], padded) + convert x ---
  prep_wg_zero<<<(512 * 512 + 255) / 256, 256, 0, stream>>>(Wgt, gbp, gb, G);
  prep_wg_scatter<<<(G * Kmax + 255) / 256, 256, 0, stream>>>(Wgt, gw, mask, idx, G, Kmax);
  prep_transpose<<<(512 * 128 + 255) / 256, 256, 0, stream>>>(w1,  w1t,  G,   128, 512, 128);
  prep_transpose<<<(128 * 128 + 255) / 256, 256, 0, stream>>>(w2,  w2t,  128, 32,  128, 128);
  prep_transpose<<<(32  * 128 + 255) / 256, 256, 0, stream>>>(dw1, dw1t, 32,  128, 32,  128);
  prep_transpose<<<(128 * 256 + 255) / 256, 256, 0, stream>>>(dw2, dw2t, 128, 256, 128, 256);
  prep_transpose<<<(256 * 512 + 255) / 256, 256, 0, stream>>>(dw3, dw3t, 256, 512, 256, 512);
  prep_cvt_bf16<<<(BATCH * 512 / 4 + 255) / 256, 256, 0, stream>>>(x, xb, BATCH * 512 / 4);

  // --- GEMM chain (block tile 128x128) ---
  gemm_tdm<__bf16, ACT_RELU><<<dim3(MT, 4), 256, LDS_TOTAL, stream>>>(xb, Wgt,  gbp, h1, 512, 512);
  gemm_tdm<__bf16, ACT_RELU><<<dim3(MT, 1), 256, LDS_TOTAL, stream>>>(h1, w1t,  b1,  h2, 512, 128);
  gemm_tdm<__bf16, ACT_RELU><<<dim3(MT, 1), 256, LDS_TOTAL, stream>>>(h2, w2t,  b2,  h3, 128, 32);
  gemm_tdm<__bf16, ACT_RELU><<<dim3(MT, 1), 256, LDS_TOTAL, stream>>>(h3, dw1t, db1, h4, 32,  128);
  gemm_tdm<__bf16, ACT_RELU><<<dim3(MT, 2), 256, LDS_TOTAL, stream>>>(h4, dw2t, db2, h5, 128, 256);
  gemm_tdm<float,  ACT_SIG ><<<dim3(MT, 4), 256, LDS_TOTAL, stream>>>(h5, dw3t, db3, (float*)d_out, 256, 512);
}
